// Attention_52158082842931
// MI455X (gfx1250) — matrix operand order repaired and checked by reading, not compiled, not run on hardware
//
#include <hip/hip_runtime.h>

// Problem constants (fixed by the reference)
#define BD  8
#define SQ  2048
#define SKV 2048
#define DD  512
#define HH  512

// Register-blocking factors: each wave owns a (16*MT) x (16*NT) macro-tile
#define MT  4
#define NT  2

typedef __attribute__((ext_vector_type(16))) __bf16        v16bf;
typedef __attribute__((ext_vector_type(8)))  float         v8f;
typedef __attribute__((ext_vector_type(4)))  float         f32x4;
typedef __attribute__((ext_vector_type(4)))  unsigned int  u32x4;

// 16 bf16 elements = one WMMA A/B operand per lane (32 bytes)
union BF16Tile {
    v16bf          v;
    unsigned short h[16];
    unsigned int   w[8];
    u32x4          q[2];
};

// float -> bf16 bits, round-to-nearest (ties away), 2 VALU ops
__device__ __forceinline__ unsigned short f2bf(float f) {
    return (unsigned short)((__float_as_uint(f) + 0x8000u) >> 16);
}

// Pack two floats into {bf16(hi),bf16(lo)} with one v_perm_b32 (+2 rounding adds)
__device__ __forceinline__ unsigned int pkbf(float lo, float hi) {
    unsigned int ul = __float_as_uint(lo) + 0x8000u;
    unsigned int uh = __float_as_uint(hi) + 0x8000u;
    // result bytes: [1:0] = ul bytes 3:2, [3:2] = uh bytes 3:2
    return __builtin_amdgcn_perm(uh, ul, 0x07060302u);
}

// Fill 8 consecutive bf16 slots (4 dwords) from two float4 chunks
__device__ __forceinline__ void fill8(BF16Tile& t, int wbase, f32x4 a, f32x4 b) {
    t.w[wbase + 0] = pkbf(a.x, a.y);
    t.w[wbase + 1] = pkbf(a.z, a.w);
    t.w[wbase + 2] = pkbf(b.x, b.y);
    t.w[wbase + 3] = pkbf(b.z, b.w);
}

// Load a fragment: fp32 source row, convert to bf16 (A-operand layout)
__device__ __forceinline__ void load_frag_f32(BF16Tile& t, const float* src) {
    f32x4 f0 = *(const f32x4*)(src);
    f32x4 f1 = *(const f32x4*)(src + 4);
    f32x4 f2 = *(const f32x4*)(src + 16);
    f32x4 f3 = *(const f32x4*)(src + 20);
    fill8(t, 0, f0, f1);
    fill8(t, 4, f2, f3);
}

// Load a fragment: bf16 source, A-operand chunking {base..+7, base+16..+23}
__device__ __forceinline__ void load_fragA_bf16(BF16Tile& t, const unsigned short* src) {
    t.q[0] = *(const u32x4*)(src);
    t.q[1] = *(const u32x4*)(src + 16);
}

// Load a fragment: bf16 source, B-operand chunking {base..+7, base+16..+23}
__device__ __forceinline__ void load_fragB_bf16(BF16Tile& t, const unsigned short* src) {
    t.q[0] = *(const u32x4*)(src);
    t.q[1] = *(const u32x4*)(src + 16);
}

__device__ __forceinline__ v8f wmma_bf16(const BF16Tile& a, const BF16Tile& b, v8f c) {
    return __builtin_amdgcn_wmma_f32_16x16x32_bf16(
        /*neg_a=*/false, a.v, /*neg_b=*/false, b.v,
        /*c_mod=*/(short)0, c, /*reuse_a=*/false, /*reuse_b=*/false);
}

// ---------------------------------------------------------------------------
// Kernel 0: transpose+convert the projection weights once.
//   Wt[g][n][k] (bf16, K-contiguous per output column) = W_g[k][n] (fp32)
// ---------------------------------------------------------------------------
__global__ __launch_bounds__(256)
void wtrans_kernel(const float* __restrict__ Wq, const float* __restrict__ Wk,
                   const float* __restrict__ Wv, unsigned short* __restrict__ wt) {
    const int g = blockIdx.z;
    const float* W = (g == 0) ? Wq : (g == 1) ? Wk : Wv;
    const int n = blockIdx.x * 256 + threadIdx.x;   // coalesced read along n
    const int k = blockIdx.y;                        // 0..DD-1
    wt[((size_t)g * HH + n) * DD + k] = f2bf(W[(size_t)k * HH + n]);
}

// ---------------------------------------------------------------------------
// Kernel 1: q/k/v projections.  X[BD*S, DD] @ W[DD, HH] + bias.
//   g==0 -> q (bf16 row-major), g==1 -> k (bf16 row-major),
//   g==2 -> v stored TRANSPOSED as vtb[B][H][S] (bf16) for the context GEMM.
// 4x2 register blocking: 8 WMMA per K-step; A converted fp32->bf16 via v_perm.
// ---------------------------------------------------------------------------
__global__ __launch_bounds__(128)
void proj_kernel(const float* __restrict__ q_in, const float* __restrict__ k_in,
                 const float* __restrict__ v_in,
                 const float* __restrict__ bq, const float* __restrict__ bk,
                 const float* __restrict__ bv,
                 const unsigned short* __restrict__ wt,
                 unsigned short* __restrict__ qb, unsigned short* __restrict__ kb,
                 unsigned short* __restrict__ vtb) {
    const int g = blockIdx.z;
    const float* X    = (g == 0) ? q_in : (g == 1) ? k_in : v_in;
    const float* bias = (g == 0) ? bq   : (g == 1) ? bk   : bv;

    const int lane = threadIdx.x;            // 0..31
    const int half = lane >> 4;              // 0 | 1
    const int l    = lane & 15;
    const int r0   = blockIdx.x * (16 * MT);                        // rows over B*S
    const int n0   = (blockIdx.y * 4 + threadIdx.y) * (16 * NT);    // output cols

    const float*          arow = X  + (size_t)(r0 + l) * DD + half * 8;
    const unsigned short* wrow = wt + ((size_t)g * HH + n0 + l) * DD + half * 8;

    v8f acc[MT][NT] = {};
    for (int k0 = 0; k0 < DD; k0 += 32) {
        BF16Tile a[MT], bm[NT];
#pragma unroll
        for (int mt = 0; mt < MT; ++mt)
            load_frag_f32(a[mt], arow + (size_t)mt * 16 * DD + k0);
#pragma unroll
        for (int nt = 0; nt < NT; ++nt)
            load_fragB_bf16(bm[nt], wrow + (size_t)nt * 16 * DD + k0);
#pragma unroll
        for (int mt = 0; mt < MT; ++mt)
#pragma unroll
            for (int nt = 0; nt < NT; ++nt)
                acc[mt][nt] = wmma_bf16(a[mt], bm[nt], acc[mt][nt]);
    }

    float bn[NT];
#pragma unroll
    for (int nt = 0; nt < NT; ++nt) bn[nt] = bias[n0 + nt * 16 + l];

    if (g == 2) {
#pragma unroll
        for (int mt = 0; mt < MT; ++mt)
#pragma unroll
            for (int nt = 0; nt < NT; ++nt)
#pragma unroll
                for (int r = 0; r < 8; ++r) {
                    const int m = r0 + mt * 16 + r + half * 8;
                    const int n = n0 + nt * 16 + l;
                    vtb[((size_t)(m >> 11) * HH + n) * SKV + (m & (SKV - 1))] =
                        f2bf(acc[mt][nt][r] + bn[nt]);
                }
    } else {
        unsigned short* dst = g ? kb : qb;
#pragma unroll
        for (int mt = 0; mt < MT; ++mt)
#pragma unroll
            for (int nt = 0; nt < NT; ++nt)
#pragma unroll
                for (int r = 0; r < 8; ++r) {
                    const int m = r0 + mt * 16 + r + half * 8;
                    const int n = n0 + nt * 16 + l;
                    dst[(size_t)m * HH + n] = f2bf(acc[mt][nt][r] + bn[nt]);
                }
    }
}

// ---------------------------------------------------------------------------
// Kernel 2: scores[b][i][j] = sum_h qb[b][i][h] * kb[b][j][h]
// Pure bf16 loads; raw fp32 scores written into the weights region of d_out.
// ---------------------------------------------------------------------------
__global__ __launch_bounds__(128)
void scores_kernel(const unsigned short* __restrict__ qb,
                   const unsigned short* __restrict__ kb,
                   float* __restrict__ wout) {
    const int lane = threadIdx.x;
    const int half = lane >> 4;
    const int l    = lane & 15;
    const int b    = blockIdx.z;
    const int i0   = blockIdx.x * (16 * MT);
    const int j0   = (blockIdx.y * 4 + threadIdx.y) * (16 * NT);

    const unsigned short* qrow = qb + ((size_t)b * SQ  + i0 + l) * HH + half * 8;
    const unsigned short* krow = kb + ((size_t)b * SKV + j0 + l) * HH + half * 8;

    v8f acc[MT][NT] = {};
    for (int k0 = 0; k0 < HH; k0 += 32) {
        BF16Tile a[MT], bm[NT];
#pragma unroll
        for (int mt = 0; mt < MT; ++mt)
            load_fragA_bf16(a[mt], qrow + (size_t)mt * 16 * HH + k0);
#pragma unroll
        for (int nt = 0; nt < NT; ++nt)
            load_fragB_bf16(bm[nt], krow + (size_t)nt * 16 * HH + k0);
#pragma unroll
        for (int mt = 0; mt < MT; ++mt)
#pragma unroll
            for (int nt = 0; nt < NT; ++nt)
                acc[mt][nt] = wmma_bf16(a[mt], bm[nt], acc[mt][nt]);
    }

    float* dst = wout + (size_t)b * SQ * SKV;
#pragma unroll
    for (int mt = 0; mt < MT; ++mt)
#pragma unroll
        for (int nt = 0; nt < NT; ++nt)
#pragma unroll
            for (int r = 0; r < 8; ++r)
                dst[(size_t)(i0 + mt * 16 + r + half * 8) * SKV +
                    (j0 + nt * 16 + l)] = acc[mt][nt][r];
}

// ---------------------------------------------------------------------------
// Kernel 3: in-place row softmax over SKV=2048 (fp32 in d_out), additionally
// emitting a bf16 copy of the weights for the context GEMM's A-operand.
// ---------------------------------------------------------------------------
__global__ __launch_bounds__(256)
void softmax_kernel(float* __restrict__ w, unsigned short* __restrict__ pb) {
    const size_t row = blockIdx.x;
    float*          p  = w  + row * SKV;
    unsigned short* po = pb + row * SKV;
    const int t = threadIdx.x;
    __shared__ float red[256];

    float vals[8];
    float m = -3.402823466e38f;
#pragma unroll
    for (int u = 0; u < 8; ++u) {
        vals[u] = p[t + u * 256];
        m = fmaxf(m, vals[u]);
    }
    red[t] = m; __syncthreads();
    for (int s = 128; s > 0; s >>= 1) {
        if (t < s) red[t] = fmaxf(red[t], red[t + s]);
        __syncthreads();
    }
    m = red[0]; __syncthreads();

    float sum = 0.f;
#pragma unroll
    for (int u = 0; u < 8; ++u) {
        vals[u] = __expf(vals[u] - m);
        sum += vals[u];
    }
    red[t] = sum; __syncthreads();
    for (int s = 128; s > 0; s >>= 1) {
        if (t < s) red[t] += red[t + s];
        __syncthreads();
    }
    const float inv = 1.0f / red[0];
#pragma unroll
    for (int u = 0; u < 8; ++u) {
        const float o = vals[u] * inv;
        p[t + u * 256]  = o;
        po[t + u * 256] = f2bf(o);
    }
}

// ---------------------------------------------------------------------------
// Kernel 4: context[b][i][h] = sum_j P[b][i][j] * v[b][j][h]
// Pure bf16 loads: A = bf16 probs from softmax, B = vT. 4x2 blocked.
// ---------------------------------------------------------------------------
__global__ __launch_bounds__(128)
void context_kernel(const unsigned short* __restrict__ pb,
                    const unsigned short* __restrict__ vtb,
                    float* __restrict__ out) {
    const int lane = threadIdx.x;
    const int half = lane >> 4;
    const int l    = lane & 15;
    const int b    = blockIdx.z;
    const int i0   = blockIdx.x * (16 * MT);
    const int h0   = (blockIdx.y * 4 + threadIdx.y) * (16 * NT);

    const unsigned short* prow = pb  + ((size_t)b * SQ + i0 + l) * SKV + half * 8;
    const unsigned short* vrow = vtb + ((size_t)b * HH + h0 + l) * SKV + half * 8;

    v8f acc[MT][NT] = {};
    for (int j0 = 0; j0 < SKV; j0 += 32) {
        BF16Tile a[MT], bm[NT];
#pragma unroll
        for (int mt = 0; mt < MT; ++mt)
            load_fragA_bf16(a[mt], prow + (size_t)mt * 16 * SKV + j0);
#pragma unroll
        for (int nt = 0; nt < NT; ++nt)
            load_fragB_bf16(bm[nt], vrow + (size_t)nt * 16 * SKV + j0);
#pragma unroll
        for (int mt = 0; mt < MT; ++mt)
#pragma unroll
            for (int nt = 0; nt < NT; ++nt)
                acc[mt][nt] = wmma_bf16(a[mt], bm[nt], acc[mt][nt]);
    }

#pragma unroll
    for (int mt = 0; mt < MT; ++mt)
#pragma unroll
        for (int nt = 0; nt < NT; ++nt)
#pragma unroll
            for (int r = 0; r < 8; ++r)
                out[((size_t)b * SQ + i0 + mt * 16 + r + half * 8) * HH +
                    (h0 + nt * 16 + l)] = acc[mt][nt][r];
}

// ---------------------------------------------------------------------------
extern "C" void kernel_launch(void* const* d_in, const int* in_sizes, int n_in,
                              void* d_out, int out_size, void* d_ws, size_t ws_size,
                              hipStream_t stream) {
    (void)in_sizes; (void)n_in; (void)out_size; (void)ws_size;

    const float* query = (const float*)d_in[0];
    const float* key_  = (const float*)d_in[1];
    const float* value = (const float*)d_in[2];
    const float* Wq    = (const float*)d_in[3];
    const float* bq    = (const float*)d_in[4];
    const float* Wk    = (const float*)d_in[5];
    const float* bk    = (const float*)d_in[6];
    const float* Wv    = (const float*)d_in[7];
    const float* bv    = (const float*)d_in[8];

    float* ctx = (float*)d_out;                         // [B, SQ, HH]
    float* wts = ctx + (size_t)BD * SQ * HH;            // [B, SQ, SKV]

    unsigned short* qb  = (unsigned short*)d_ws;        // [B*SQ,  HH]  bf16
    unsigned short* kb  = qb  + (size_t)BD * SQ  * HH;  // [B*SKV, HH]  bf16
    unsigned short* vtb = kb  + (size_t)BD * SKV * HH;  // [B, HH, SKV] bf16
    unsigned short* wt  = vtb + (size_t)BD * HH  * SKV; // [3, HH, DD]  bf16
    unsigned short* pb  = wt  + (size_t)3  * HH  * DD;  // [B*SQ, SKV]  bf16

    const dim3 blk(32, 4);  // 4 x wave32, one 64x32 macro-tile per wave

    wtrans_kernel<<<dim3(HH / 256, DD, 3), dim3(256), 0, stream>>>(Wq, Wk, Wv, wt);

    proj_kernel<<<dim3((BD * SQ) / (16 * MT), (HH / (16 * NT)) / 4, 3),
                  blk, 0, stream>>>(query, key_, value, bq, bk, bv, wt, qb, kb, vtb);

    scores_kernel<<<dim3(SQ / (16 * MT), (SKV / (16 * NT)) / 4, BD),
                    blk, 0, stream>>>(qb, kb, wts);

    softmax_kernel<<<dim3(BD * SQ), 256, 0, stream>>>(wts, pb);

    context_kernel<<<dim3(SQ / (16 * MT), (HH / (16 * NT)) / 4, BD),
                     blk, 0, stream>>>(pb, vtb, ctx);
}
